// TimeDecayMultiheadAttention_31293131719464
// MI455X (gfx1250) — matrix operand order repaired and checked by reading, not compiled, not run on hardware
//
#include <hip/hip_runtime.h>
#include <cstdint>
#include <cstddef>

// ---------------------------------------------------------------------------
// TimeDecayMultiheadAttention for MI455X (gfx1250, wave32, WMMA bf16)
// T=2048, B=4, D=1024, H=16, HD=64
// ---------------------------------------------------------------------------

#define T_SEQ 2048
#define BATCH 4
#define DMODEL 1024
#define NHEAD 16
#define HDIM 64

typedef __attribute__((ext_vector_type(16))) __bf16 v16bf;
typedef __attribute__((ext_vector_type(8)))  float  v8f;

__device__ __forceinline__ unsigned int f2bf(float a) {
  union { float f; unsigned int u; } x; x.f = a;
  // round-to-nearest-even to bf16
  return (x.u + 0x7FFFu + ((x.u >> 16) & 1u)) >> 16;
}
__device__ __forceinline__ unsigned int pk2bf(float lo, float hi) {
  return f2bf(lo) | (f2bf(hi) << 16);
}

union Frag {             // one WMMA 16-bit operand: 16 bf16 per lane (8 VGPRs)
  v16bf v;
  uint4 q[2];
  unsigned int u[8];
};

// ---------------------------------------------------------------------------
// GEMM: Y = X @ W^T + b, X:(8192,1024), W:(1024,1024) row-major.
// Block: 256 thr = 8 waves; tile 128(M) x 64(N); k-step 32.
// E==0: Q proj  -> bf16 Qp[(b*H+h)*T + t][hd], scaled by 0.125*log2(e)
// E==1: K proj  -> bf16 Kp  (same layout)
// E==2: V proj  -> bf16 Vt[(b*H+h)*HD + hd][t]   (transposed)
// E==3: out proj: X is bf16 ctx; Y = fp32 out * time_decay
// ---------------------------------------------------------------------------
template <int E>
__global__ __launch_bounds__(256) void gemm_kernel(
    const void* __restrict__ Xv, const float* __restrict__ W,
    const float* __restrict__ bias, void* __restrict__ Yv,
    const float* __restrict__ td) {
  __shared__ unsigned short Xs[128 * 40];  // 32 bf16 + pad per row
  __shared__ unsigned short Ws[64 * 40];

  const int tid  = threadIdx.x;
  const int lane = tid & 31;
  const int wave = tid >> 5;
  const int qr   = lane & 15;   // M/N index within 16x16 tile
  const int hh   = lane >> 4;   // half-wave
  const int rowbase = blockIdx.x * 128;
  const int colbase = blockIdx.y * 64;

  v8f acc[4];
#pragma unroll
  for (int g = 0; g < 4; ++g)
#pragma unroll
    for (int i = 0; i < 8; ++i) acc[g][i] = 0.0f;

  float bvals[4];
#pragma unroll
  for (int g = 0; g < 4; ++g) bvals[g] = bias[colbase + g * 16 + qr];

#pragma unroll 1
  for (int kk = 0; kk < DMODEL; kk += 32) {
    // ---- stage X tile (128 x 32) into LDS as bf16 ----
    if (E == 3) {
      const unsigned short* X = (const unsigned short*)Xv;
#pragma unroll
      for (int qi = 0; qi < 2; ++qi) {
        int idx = tid + 256 * qi;          // 0..511 uint4 (8 bf16 each)
        int row = idx >> 2;
        int kq  = (idx & 3) * 8;
        uint4 d = *(const uint4*)(X + (size_t)(rowbase + row) * DMODEL + kk + kq);
        *(uint4*)(&Xs[row * 40 + kq]) = d;
      }
    } else {
      const float* X = (const float*)Xv;
#pragma unroll
      for (int qi = 0; qi < 4; ++qi) {
        int idx = tid + 256 * qi;          // 0..1023 float4
        int row = idx >> 3;
        int kq  = (idx & 7) * 4;
        float4 d = *(const float4*)(X + (size_t)(rowbase + row) * DMODEL + kk + kq);
        uint2 p;
        p.x = pk2bf(d.x, d.y);
        p.y = pk2bf(d.z, d.w);
        *(uint2*)(&Xs[row * 40 + kq]) = p;
      }
    }
    // ---- stage W tile (64 x 32) into LDS as bf16 ----
#pragma unroll
    for (int qi = 0; qi < 2; ++qi) {
      int idx = tid + 256 * qi;            // 0..511 float4
      int row = idx >> 3;
      int kq  = (idx & 7) * 4;
      float4 d = *(const float4*)(W + (size_t)(colbase + row) * DMODEL + kk + kq);
      uint2 p;
      p.x = pk2bf(d.x, d.y);
      p.y = pk2bf(d.z, d.w);
      *(uint2*)(&Ws[row * 40 + kq]) = p;
    }
    __syncthreads();

    // A fragment: row = wave*16 + qr; k-chunks at hh*8 and hh*8+16
    Frag a;
    const int m = wave * 16 + qr;
    a.q[0] = *(const uint4*)(&Xs[m * 40 + hh * 8]);
    a.q[1] = *(const uint4*)(&Xs[m * 40 + hh * 8 + 16]);
#pragma unroll
    for (int g = 0; g < 4; ++g) {
      // B fragment: n = g*16 + qr; k-chunks at hh*8 and hh*8+16
      Frag b;
      const int n = g * 16 + qr;
      b.q[0] = *(const uint4*)(&Ws[n * 40 + hh * 8]);
      b.q[1] = *(const uint4*)(&Ws[n * 40 + hh * 8 + 16]);
      acc[g] = __builtin_amdgcn_wmma_f32_16x16x32_bf16(
          false, a.v, false, b.v, (short)0, acc[g], false, false);
    }
    __syncthreads();
  }

  // ---- epilogue ----
#pragma unroll
  for (int g = 0; g < 4; ++g) {
    const int n = colbase + g * 16 + qr;
#pragma unroll
    for (int v = 0; v < 8; ++v) {
      const int r = rowbase + wave * 16 + v + 8 * hh;  // C layout: M = v + 8*hh
      float c = acc[g][v] + bvals[g];
      if (E == 0 || E == 1 || E == 2) {
        if (E == 0) c *= 0.1803368801111204f;  // (1/sqrt(64)) * log2(e)
        const int t = r >> 2, b_ = r & 3;      // r = t*BATCH + b
        const int hidx = n >> 6, hd = n & 63;
        unsigned short* Y = (unsigned short*)Yv;
        size_t off;
        if (E == 2)  // transposed V
          off = ((size_t)(b_ * NHEAD + hidx) * HDIM + hd) * T_SEQ + t;
        else
          off = ((size_t)(b_ * NHEAD + hidx) * T_SEQ + t) * HDIM + hd;
        Y[off] = (unsigned short)f2bf(c);
      } else {
        float* Y = (float*)Yv;
        const size_t off = (size_t)r * DMODEL + n;
        Y[off] = c * td[off];
      }
    }
  }
}

// ---------------------------------------------------------------------------
// Flash attention (transposed): per wave, 16 query rows; S^T = K.Q^T,
// O^T = V^T.P^T.  Q already carries 0.125*log2e, so softmax uses exp2.
// Block = 128 thr = 4 waves (64 queries); grid = (T/64, B*H).
// ---------------------------------------------------------------------------
__global__ __launch_bounds__(128) void attn_kernel(
    const unsigned short* __restrict__ Qp, const unsigned short* __restrict__ Kp,
    const unsigned short* __restrict__ Vt, unsigned short* __restrict__ ctx) {
  const int lane = threadIdx.x & 31;
  const int wave = threadIdx.x >> 5;
  const int qr   = lane & 15;  // query row within tile (N of S^T)
  const int hh   = lane >> 4;
  const int bh   = blockIdx.y;
  const int qbase = blockIdx.x * 64 + wave * 16;

  // Q^T B-fragments (hd x qrow), hd windows w=0,1: contiguous 16 bf16 per lane
  Frag Qf[2];
  {
    const unsigned short* qrow =
        Qp + ((size_t)bh * T_SEQ + qbase + qr) * HDIM;
#pragma unroll
    for (int w = 0; w < 2; ++w) {
      Qf[w].q[0] = *(const uint4*)(qrow + w * 32 + hh * 16);
      Qf[w].q[1] = *(const uint4*)(qrow + w * 32 + hh * 16 + 8);
    }
  }

  v8f o[4];
#pragma unroll
  for (int g = 0; g < 4; ++g)
#pragma unroll
    for (int i = 0; i < 8; ++i) o[g][i] = 0.0f;
  float m = -3.0e38f, l = 0.0f;

#pragma unroll 1
  for (int kb = 0; kb < T_SEQ; kb += 32) {
    // S^T tiles: keys kb+j*16..+15 (M) x 16 queries (N)
    v8f st[2];
#pragma unroll
    for (int j = 0; j < 2; ++j) {
#pragma unroll
      for (int i = 0; i < 8; ++i) st[j][i] = 0.0f;
      const unsigned short* krow =
          Kp + ((size_t)bh * T_SEQ + kb + j * 16 + qr) * HDIM;
#pragma unroll
      for (int w = 0; w < 2; ++w) {
        Frag kf;  // A fragment (key rows): contiguous 16 at hh*16 in hd window
        kf.q[0] = *(const uint4*)(krow + w * 32 + hh * 16);
        kf.q[1] = *(const uint4*)(krow + w * 32 + hh * 16 + 8);
        st[j] = __builtin_amdgcn_wmma_f32_16x16x32_bf16(
            false, kf.v, false, Qf[w].v, (short)0, st[j], false, false);
      }
    }

    // Online softmax: query row == lane group -> one cross-half shuffle
    float mloc = st[0][0];
#pragma unroll
    for (int i = 0; i < 8; ++i) {
      mloc = fmaxf(mloc, st[0][i]);
      mloc = fmaxf(mloc, st[1][i]);
    }
    mloc = fmaxf(mloc, __shfl_xor(mloc, 16, 32));
    const float mn   = fmaxf(m, mloc);
    const float corr = exp2f(m - mn);
    float p0[8], p1[8], rsum = 0.0f;
#pragma unroll
    for (int i = 0; i < 8; ++i) {
      p0[i] = exp2f(st[0][i] - mn);
      p1[i] = exp2f(st[1][i] - mn);
      rsum += p0[i] + p1[i];
    }
    rsum += __shfl_xor(rsum, 16, 32);
    l = l * corr + rsum;
    m = mn;
#pragma unroll
    for (int g = 0; g < 4; ++g)
#pragma unroll
      for (int i = 0; i < 8; ++i) o[g][i] *= corr;

    // Build P^T B-fragment (keys x qrow) from S^T C-layout via xor-16 shuffles
    float lo[8], hi[8];
#pragma unroll
    for (int i = 0; i < 8; ++i) {
      const float o0 = __shfl_xor(p0[i], 16, 32);
      const float o1 = __shfl_xor(p1[i], 16, 32);
      lo[i] = hh ? o1 : p0[i];  // keys hh*16 + 0..7
      hi[i] = hh ? p1[i] : o0;  // keys hh*16 + 8..15
    }
    Frag pf;
#pragma unroll
    for (int i = 0; i < 4; ++i) {
      pf.u[i]     = pk2bf(lo[2 * i], lo[2 * i + 1]);
      pf.u[4 + i] = pk2bf(hi[2 * i], hi[2 * i + 1]);
    }

    // O^T += V^T @ P^T : A = Vt rows (hd), k = keys
#pragma unroll
    for (int g = 0; g < 4; ++g) {
      const unsigned short* vrow =
          Vt + ((size_t)bh * HDIM + g * 16 + qr) * T_SEQ + kb;
      Frag vf;
      vf.q[0] = *(const uint4*)(vrow + hh * 16);
      vf.q[1] = *(const uint4*)(vrow + hh * 16 + 8);
      o[g] = __builtin_amdgcn_wmma_f32_16x16x32_bf16(
          false, vf.v, false, pf.v, (short)0, o[g], false, false);
    }
  }

  // Finalize: ctx[(t*B + b)*D + head*64 + hd], hd = g*16 + hh*8 + v
  const float inv = 1.0f / l;
  const int t  = qbase + qr;
  const int b_ = bh / NHEAD;
  const int d0 = (bh % NHEAD) * HDIM;
#pragma unroll
  for (int g = 0; g < 4; ++g) {
    uint4 w;
    unsigned int* wp = (unsigned int*)&w;
#pragma unroll
    for (int i = 0; i < 4; ++i)
      wp[i] = pk2bf(o[g][2 * i] * inv, o[g][2 * i + 1] * inv);
    *(uint4*)(ctx + (size_t)(t * BATCH + b_) * DMODEL + d0 + g * 16 + hh * 8) = w;
  }
}

// ---------------------------------------------------------------------------
extern "C" void kernel_launch(void* const* d_in, const int* in_sizes, int n_in,
                              void* d_out, int out_size, void* d_ws,
                              size_t ws_size, hipStream_t stream) {
  const float* query = (const float*)d_in[0];
  const float* key_  = (const float*)d_in[1];
  const float* value = (const float*)d_in[2];
  const float* td    = (const float*)d_in[3];
  const float* Wq = (const float*)d_in[4];
  const float* bq = (const float*)d_in[5];
  const float* Wk = (const float*)d_in[6];
  const float* bk = (const float*)d_in[7];
  const float* Wv = (const float*)d_in[8];
  const float* bv = (const float*)d_in[9];
  const float* Wo = (const float*)d_in[10];
  const float* bo = (const float*)d_in[11];
  float* out = (float*)d_out;

  const size_t NE = (size_t)T_SEQ * BATCH * DMODEL;  // 8M elements
  unsigned short* ws  = (unsigned short*)d_ws;
  unsigned short* Qp  = ws;            // bf16, (B*H, T, HD)
  unsigned short* Kp  = ws + NE;       // bf16, (B*H, T, HD)
  unsigned short* Vt  = ws + 2 * NE;   // bf16, (B*H, HD, T)
  unsigned short* ctx = ws + 3 * NE;   // bf16, (T, B, D)

  dim3 gg(T_SEQ * BATCH / 128, DMODEL / 64);  // (64, 16)
  gemm_kernel<0><<<gg, 256, 0, stream>>>(query, Wq, bq, Qp, nullptr);
  gemm_kernel<1><<<gg, 256, 0, stream>>>(key_, Wk, bk, Kp, nullptr);
  gemm_kernel<2><<<gg, 256, 0, stream>>>(value, Wv, bv, Vt, nullptr);

  dim3 ga(T_SEQ / 64, BATCH * NHEAD);  // (32, 64)
  attn_kernel<<<ga, 128, 0, stream>>>(Qp, Kp, Vt, ctx);

  gemm_kernel<3><<<gg, 256, 0, stream>>>(ctx, Wo, bo, out, td);
}
